// BiLSTM_CRF_80204219286414
// MI455X (gfx1250) — matrix operand order repaired and checked by reading, not compiled, not run on hardware
//
#include <hip/hip_runtime.h>

// ---------------- types & helpers ----------------
typedef __bf16 v16bf __attribute__((ext_vector_type(16)));
typedef float  v8f   __attribute__((ext_vector_type(8)));
typedef unsigned int v4u __attribute__((ext_vector_type(4)));
typedef unsigned int v2u __attribute__((ext_vector_type(2)));

union Frag { v16bf bf; v4u q[2]; };

constexpr int Tn = 512, Bn = 64, En = 300, Ep = 320, Hh = 256, NG = 1024;
constexpr int Kc = 20, TBEGIN = 18, TEND = 19;
constexpr float NEGV = -10000.0f;

static __device__ __forceinline__ unsigned short f2bf(float f) {
  unsigned u = __float_as_uint(f);
  u += 0x7FFFu + ((u >> 16) & 1u);           // round-to-nearest-even
  return (unsigned short)(u >> 16);
}
static __device__ __forceinline__ float bf2f(unsigned short h) {
  return __uint_as_float(((unsigned)h) << 16);
}
static __device__ __forceinline__ float sigf(float x) {
  return 1.0f / (1.0f + __expf(-x));
}

// A fragment: 16x32 bf16, row-major source [M, ld]; per ISA layout
// lanes 0-15 hold M=lane: V0..3 = K 0..7, V4..7 = K 16..23; lanes 16-31: K+8.
static __device__ __forceinline__ v16bf frag_a(const unsigned short* base, int ld, int lane) {
  int r = lane & 15, hi = lane >> 4;
  const unsigned short* pp = base + r * ld + (hi << 3);
  Frag f;
  f.q[0] = *(const v4u*)(pp);
  f.q[1] = *(const v4u*)(pp + 16);
  return f.bf;
}
// B fragment: 32x16 bf16 where B[k][n] = W[n][k], W row-major [N, ld].
// lanes 0-15 hold column n: V0..3 = K 0..7, V4..7 = K 16..23; lanes 16-31: K+8.
static __device__ __forceinline__ v16bf frag_b(const unsigned short* base, int ld, int lane) {
  int n = lane & 15, hi = lane >> 4;
  const unsigned short* pp = base + n * ld + (hi << 3);
  Frag f;
  f.q[0] = *(const v4u*)(pp);
  f.q[1] = *(const v4u*)(pp + 16);
  return f.bf;
}
static __device__ __forceinline__ v8f wmma_bf16(v16bf a, v16bf b, v8f c) {
  return __builtin_amdgcn_wmma_f32_16x16x32_bf16(false, a, false, b, (short)0, c, false, false);
}

// ---------------- kernel 1: f32 -> bf16 weight convert with K/row padding ----------------
__global__ void cvt_pad_kernel(const float* __restrict__ src, unsigned short* __restrict__ dst,
                               int rows, int srcRows, int kdst, int ksrc) {
  int i = blockIdx.x * 256 + threadIdx.x;
  if (i >= rows * kdst) return;
  int r = i / kdst, k = i - r * kdst;
  float v = (r < srcRows && k < ksrc) ? src[r * ksrc + k] : 0.0f;
  dst[i] = f2bf(v);
}

// ---------------- kernel 2: embedding gather -> x bf16 [T*B, 320] ----------------
__global__ void embed_kernel(const int* __restrict__ sent, const float* __restrict__ table,
                             unsigned short* __restrict__ x) {
  int i = blockIdx.x * 256 + threadIdx.x;                  // < T*B*Ep
  int e = i % Ep;
  int r = i / Ep;                                          // r = t*B + b
  int b = r % Bn, t = r / Bn;
  int w = sent[b * Tn + t];
  float v = (e < En) ? table[(size_t)w * En + e] : 0.0f;
  x[i] = f2bf(v);
}

// ---------------- kernel 3: time-parallel input projection ----------------
// out[row][col*4 + gate] = (x @ Wih^T)[row][gate*256+col]  (gate-interleaved layout so the
// recurrent kernel's pointwise stage reads its 4 gate pre-activations with one b64 load)
__global__ __launch_bounds__(512) void xproj_kernel(const unsigned short* __restrict__ x,
                                                    const unsigned short* __restrict__ wih,
                                                    unsigned short* __restrict__ out) {
  int row0 = blockIdx.x * 16;
  int lane = threadIdx.x & 31, w = threadIdx.x >> 5, hi = lane >> 4, n = lane & 15;
  v8f acc[4];
#pragma unroll
  for (int j = 0; j < 4; ++j)
#pragma unroll
    for (int v = 0; v < 8; ++v) acc[j][v] = 0.0f;
#pragma unroll
  for (int kk = 0; kk < 10; ++kk) {
    v16bf a = frag_a(x + (size_t)row0 * Ep + 32 * kk, Ep, lane);
#pragma unroll
    for (int j = 0; j < 4; ++j) {
      v16bf bm = frag_b(wih + (size_t)(16 * (4 * w + j)) * Ep + 32 * kk, Ep, lane);
      acc[j] = wmma_bf16(a, bm, acc[j]);
    }
  }
#pragma unroll
  for (int j = 0; j < 4; ++j) {
    int ncol = 16 * (4 * w + j) + n;     // 0..1023 gate-major column
    int gate = ncol >> 8, cc = ncol & 255;
#pragma unroll
    for (int v = 0; v < 8; ++v) {
      int m = v + (hi << 3);
      out[(size_t)(row0 + m) * NG + cc * 4 + gate] = f2bf(acc[j][v]);
    }
  }
}

// ---------------- kernel 4: persistent BiLSTM recurrence (1 WG per direction) ----------------
// Wave w owns hidden columns [16w, 16w+16). c state in registers, h ping-pong in LDS.
// M-tiles processed in pairs with Whh B-fragments hoisted per k-step (2x reuse).
__global__ __launch_bounds__(512) void lstm_kernel(const unsigned short* __restrict__ xpf,
                                                   const unsigned short* __restrict__ xpb,
                                                   const unsigned short* __restrict__ whhf,
                                                   const unsigned short* __restrict__ whhb,
                                                   const float* __restrict__ bf_,
                                                   const float* __restrict__ bb_,
                                                   const float* __restrict__ h0,
                                                   const float* __restrict__ c0,
                                                   unsigned short* __restrict__ hsf,
                                                   unsigned short* __restrict__ hsb) {
  __shared__ unsigned short hbuf[2][Bn * Hh];  // 64 KB, ping-pong
  int dir = blockIdx.x;
  const unsigned short* xp  = dir ? xpb  : xpf;
  const unsigned short* whh = dir ? whhb : whhf;
  const float*          bia = dir ? bb_  : bf_;
  unsigned short*       hs  = dir ? hsb  : hsf;

  int tid = threadIdx.x, w = tid >> 5, lane = tid & 31;
  int hi = lane >> 4;
  int col = (w << 4) + (lane & 15);           // owned hidden column

  for (int i = tid; i < Bn * Hh; i += 512) hbuf[0][i] = f2bf(h0[dir * Bn * Hh + i]);

  float bi = bia[col], bfv = bia[256 + col], bg = bia[512 + col], bo = bia[768 + col];

  float creg[4][8];
#pragma unroll
  for (int mt = 0; mt < 4; ++mt)
#pragma unroll
    for (int v = 0; v < 8; ++v) {
      int m = (mt << 4) + v + (hi << 3);
      creg[mt][v] = c0[dir * Bn * Hh + m * Hh + col];
    }
  __syncthreads();

  int cur = 0;
  for (int s = 0; s < Tn; ++s) {
    int xrow = dir ? (Tn - 1 - s) : s;        // both compute index and store index
    const unsigned short* hc = hbuf[cur];
    unsigned short*       hn = hbuf[cur ^ 1];

    if (s + 1 < Tn) {                          // prefetch next step's gate inputs (L2)
      int nr = dir ? (Tn - 2 - s) : (s + 1);
      __builtin_prefetch(xp + (size_t)(nr * Bn) * NG + tid * 128, 0, 1);
    }

#pragma unroll
    for (int mp = 0; mp < 2; ++mp) {           // M-tile pair {2mp, 2mp+1}
      v8f acc[2][4];
#pragma unroll
      for (int m2 = 0; m2 < 2; ++m2)
#pragma unroll
        for (int g = 0; g < 4; ++g)
#pragma unroll
          for (int v = 0; v < 8; ++v) acc[m2][g][v] = 0.0f;

      // gates += h @ Whh^T : K = 256, 8 WMMA k-steps; B fragments hoisted per k-step
#pragma unroll
      for (int kk = 0; kk < 8; ++kk) {
        v16bf bm[4];
#pragma unroll
        for (int g = 0; g < 4; ++g)
          bm[g] = frag_b(whh + (size_t)(g * 256 + (w << 4)) * Hh + 32 * kk, Hh, lane);
#pragma unroll
        for (int m2 = 0; m2 < 2; ++m2) {
          v16bf a = frag_a(hc + ((mp * 2 + m2) << 4) * Hh + 32 * kk, Hh, lane);
#pragma unroll
          for (int g = 0; g < 4; ++g) acc[m2][g] = wmma_bf16(a, bm[g], acc[m2][g]);
        }
      }
      // pointwise LSTM update, all in registers; gate inputs arrive as one b64 per element
#pragma unroll
      for (int m2 = 0; m2 < 2; ++m2) {
        int mt = mp * 2 + m2;
#pragma unroll
        for (int v = 0; v < 8; ++v) {
          int m = (mt << 4) + v + (hi << 3);
          const unsigned short* xr = xp + (size_t)(xrow * Bn + m) * NG + (size_t)col * 4;
          v2u xq = *(const v2u*)xr;            // {f:i, o:g} packed bf16 pairs
          float xi = bf2f((unsigned short)(xq[0] & 0xffffu));
          float xf = bf2f((unsigned short)(xq[0] >> 16));
          float xg = bf2f((unsigned short)(xq[1] & 0xffffu));
          float xo = bf2f((unsigned short)(xq[1] >> 16));
          float iv = sigf(acc[m2][0][v] + xi + bi);
          float fv = sigf(acc[m2][1][v] + xf + bfv);
          float gv = tanhf(acc[m2][2][v] + xg + bg);
          float ov = sigf(acc[m2][3][v] + xo + bo);
          float c  = fv * creg[mt][v] + iv * gv;
          creg[mt][v] = c;
          unsigned short h16 = f2bf(ov * tanhf(c));
          hn[m * Hh + col] = h16;
          hs[(size_t)(xrow * Bn + m) * Hh + col] = h16;
        }
      }
    }
    __syncthreads();
    cur ^= 1;
  }
}

// ---------------- kernel 5: feats = [hs_f | hs_b] @ Wout^T + b_out -> f32 [T*B, 32] ----------
__global__ __launch_bounds__(32) void feats_kernel(const unsigned short* __restrict__ hsf,
                                                   const unsigned short* __restrict__ hsb,
                                                   const unsigned short* __restrict__ woutb,
                                                   const float* __restrict__ bout,
                                                   float* __restrict__ feats) {
  int row0 = blockIdx.x * 16;
  int lane = threadIdx.x, hi = lane >> 4, n = lane & 15;
  v8f a0, a1;
#pragma unroll
  for (int v = 0; v < 8; ++v) { a0[v] = 0.0f; a1[v] = 0.0f; }
#pragma unroll
  for (int kk = 0; kk < 16; ++kk) {
    const unsigned short* ab = (kk < 8) ? (hsf + (size_t)row0 * Hh + 32 * kk)
                                        : (hsb + (size_t)row0 * Hh + 32 * (kk - 8));
    v16bf a  = frag_a(ab, Hh, lane);
    v16bf b0 = frag_b(woutb + 32 * kk, 512, lane);
    v16bf b1 = frag_b(woutb + (size_t)16 * 512 + 32 * kk, 512, lane);
    a0 = wmma_bf16(a, b0, a0);
    a1 = wmma_bf16(a, b1, a1);
  }
  float bo0 = bout[n];                                // cols 0..15 all < 20
  float bo1 = ((16 + n) < Kc) ? bout[16 + n] : 0.0f;  // cols 16..19 valid
#pragma unroll
  for (int v = 0; v < 8; ++v) {
    int m = v + (hi << 3);
    feats[(size_t)(row0 + m) * 32 + n]      = a0[v] + bo0;
    feats[(size_t)(row0 + m) * 32 + 16 + n] = a1[v] + bo1;
  }
}

// ---------------- kernel 6: CRF forward scan + gold score + mean (tiny) ----------------
__global__ __launch_bounds__(64) void crf_kernel(const float* __restrict__ feats,
                                                 const float* __restrict__ trans,
                                                 const int* __restrict__ tags,
                                                 const int* __restrict__ lens,
                                                 float* __restrict__ out) {
  __shared__ float tr[Kc * Kc];
  __shared__ float al[2][Bn][Kc];
  __shared__ float red[Bn];
  int b = threadIdx.x;
  for (int i = b; i < Kc * Kc; i += Bn) tr[i] = trans[i];
  int len = lens[b];
  for (int k = 0; k < Kc; ++k) al[0][b][k] = (k == TBEGIN) ? 0.0f : NEGV;
  __syncthreads();

  int cur = 0;
  for (int t = 0; t < Tn; ++t) {
    float mk = (t < len) ? 1.0f : 0.0f;
    const float* fr = feats + (size_t)(t * Bn + b) * 32;
    for (int nx = 0; nx < Kc; ++nx) {
      float fe = fr[nx];
      float mx = -3.4e38f;
      for (int pv = 0; pv < Kc; ++pv) {
        float sc = al[cur][b][pv] + mk * (tr[nx * Kc + pv] + fe);
        mx = fmaxf(mx, sc);
      }
      float sm = 0.0f;
      for (int pv = 0; pv < Kc; ++pv) {
        float sc = al[cur][b][pv] + mk * (tr[nx * Kc + pv] + fe);
        sm += __expf(sc - mx);
      }
      al[cur ^ 1][b][nx] = mx + __logf(sm);
    }
    cur ^= 1;
  }
  float mx = -3.4e38f;
  for (int k = 0; k < Kc; ++k) mx = fmaxf(mx, al[cur][b][k] + tr[TEND * Kc + k]);
  float sm = 0.0f;
  for (int k = 0; k < Kc; ++k) sm += __expf(al[cur][b][k] + tr[TEND * Kc + k] - mx);
  float fwd = mx + __logf(sm);

  float gold = 0.0f;
  int prev = TBEGIN;
  for (int t = 0; t < Tn; ++t) {
    int tg = tags[b * Tn + t];
    float mk = (t < len) ? 1.0f : 0.0f;
    gold += mk * (feats[(size_t)(t * Bn + b) * 32 + tg] + tr[tg * Kc + prev]);
    prev = tg;
  }
  int last = tags[b * Tn + (len - 1)];
  gold += tr[TEND * Kc + last];

  red[b] = fwd - gold;
  __syncthreads();
  if (b == 0) {
    float s = 0.0f;
    for (int i = 0; i < Bn; ++i) s += red[i];
    out[0] = s * (1.0f / (float)Bn);
  }
}

// ---------------- host launcher ----------------
extern "C" void kernel_launch(void* const* d_in, const int* in_sizes, int n_in,
                              void* d_out, int out_size, void* d_ws, size_t ws_size,
                              hipStream_t stream) {
  const int*   sentence = (const int*)d_in[0];
  const int*   tags     = (const int*)d_in[1];
  const int*   lens     = (const int*)d_in[2];
  const float* embed    = (const float*)d_in[3];
  const float* wihf     = (const float*)d_in[4];
  const float* whhf     = (const float*)d_in[5];
  const float* bf_      = (const float*)d_in[6];
  const float* wihb     = (const float*)d_in[7];
  const float* whhb     = (const float*)d_in[8];
  const float* bb_      = (const float*)d_in[9];
  const float* wout     = (const float*)d_in[10];
  const float* bout     = (const float*)d_in[11];
  const float* trans    = (const float*)d_in[12];
  const float* h0       = (const float*)d_in[13];
  const float* c0       = (const float*)d_in[14];

  char* p = (char*)d_ws;
  auto alloc = [&](size_t bytes) -> char* {
    char* r = p;
    p += (bytes + 255) & ~(size_t)255;
    return r;
  };
  unsigned short* xbf    = (unsigned short*)alloc((size_t)Tn * Bn * Ep * 2);
  unsigned short* wihf16 = (unsigned short*)alloc((size_t)NG * Ep * 2);
  unsigned short* wihb16 = (unsigned short*)alloc((size_t)NG * Ep * 2);
  unsigned short* whhf16 = (unsigned short*)alloc((size_t)NG * Hh * 2);
  unsigned short* whhb16 = (unsigned short*)alloc((size_t)NG * Hh * 2);
  unsigned short* wout16 = (unsigned short*)alloc((size_t)32 * 512 * 2);
  unsigned short* xpf    = (unsigned short*)alloc((size_t)Tn * Bn * NG * 2);
  unsigned short* xpb    = (unsigned short*)alloc((size_t)Tn * Bn * NG * 2);
  unsigned short* hsf    = (unsigned short*)alloc((size_t)Tn * Bn * Hh * 2);
  unsigned short* hsb    = (unsigned short*)alloc((size_t)Tn * Bn * Hh * 2);
  float*          feats  = (float*)alloc((size_t)Tn * Bn * 32 * 4);

  // weight precision/layout prep
  cvt_pad_kernel<<<(NG * Ep + 255) / 256, 256, 0, stream>>>(wihf, wihf16, NG, NG, Ep, En);
  cvt_pad_kernel<<<(NG * Ep + 255) / 256, 256, 0, stream>>>(wihb, wihb16, NG, NG, Ep, En);
  cvt_pad_kernel<<<(NG * Hh + 255) / 256, 256, 0, stream>>>(whhf, whhf16, NG, NG, Hh, Hh);
  cvt_pad_kernel<<<(NG * Hh + 255) / 256, 256, 0, stream>>>(whhb, whhb16, NG, NG, Hh, Hh);
  cvt_pad_kernel<<<(32 * 512 + 255) / 256, 256, 0, stream>>>(wout, wout16, 32, Kc, 512, 512);

  // embedding gather -> bf16 [T*B, 320]
  embed_kernel<<<(Tn * Bn * Ep) / 256, 256, 0, stream>>>(sentence, embed, xbf);

  // time-parallel input projections (whole-GPU WMMA GEMMs), gate-interleaved output
  xproj_kernel<<<Tn * Bn / 16, 512, 0, stream>>>(xbf, wihf16, xpf);
  xproj_kernel<<<Tn * Bn / 16, 512, 0, stream>>>(xbf, wihb16, xpb);

  // sequential recurrence: one persistent workgroup per direction
  lstm_kernel<<<2, 512, 0, stream>>>(xpf, xpb, whhf16, whhb16, bf_, bb_, h0, c0, hsf, hsb);

  // output projection
  feats_kernel<<<Tn * Bn / 16, 32, 0, stream>>>(hsf, hsb, wout16, bout, feats);

  // CRF scan + gold + mean
  crf_kernel<<<1, 64, 0, stream>>>(feats, trans, tags, lens, (float*)d_out);
}
